// ANEDeltaNet_13589276525252
// MI455X (gfx1250) — matrix operand order repaired and checked by reading, not compiled, not run on hardware
//
#include <hip/hip_runtime.h>

typedef __attribute__((ext_vector_type(16))) __bf16 v16bf;
typedef __attribute__((ext_vector_type(8)))  float  v8f;

namespace {
constexpr int B_ = 4, S_ = 1024, H_ = 2048;
constexpr int HK_ = 16, HV_ = 32, DK_ = 128, DV_ = 128, KW_ = 4;
constexpr int QKD_ = HK_ * DK_;                 // 2048
constexpr int VD_  = HV_ * DV_;                 // 4096
constexpr int CI_  = 2 * QKD_ + VD_;            // 8192
constexpr int LDN_ = CI_ + VD_ + HV_ + HV_;     // 12352 (fused projection width)
constexpr int MROWS_ = B_ * S_;                 // 4096
}

__device__ __forceinline__ unsigned short f2bf(float f) {
  unsigned u = __float_as_uint(f);
  u += 0x7FFFu + ((u >> 16) & 1u);              // round-to-nearest-even
  return (unsigned short)(u >> 16);
}
__device__ __forceinline__ float sigmoidf_(float x) { return 1.0f / (1.0f + __expf(-x)); }

// LDS byte offset of a __shared__ object (flat addr low 32 bits == wave-relative
// LDS offset per CDNA5 aperture mapping).
__device__ __forceinline__ unsigned ldsoff(const void* p) {
  return (unsigned)(unsigned long long)p;
}
// CDNA5 async copy global -> LDS, 16B per enabled lane; tracked by ASYNCcnt.
__device__ __forceinline__ void async_ld_b128(unsigned lds_byte_off, const float* gsrc) {
  asm volatile("global_load_async_to_lds_b128 %0, %1, off"
               :: "v"(lds_byte_off), "v"((unsigned long long)gsrc)
               : "memory");
}
__device__ __forceinline__ void wait_async0() {
  asm volatile("s_wait_asynccnt 0" ::: "memory");
}

// ---------------------------------------------------------------- f32 -> bf16
__global__ __launch_bounds__(256) void cvt_f32_bf16(const float* __restrict__ src,
                                                    unsigned short* __restrict__ dst,
                                                    long long n) {
  long long i = (long long)blockIdx.x * blockDim.x + threadIdx.x;
  long long stride = (long long)gridDim.x * blockDim.x;
  for (; i < n; i += stride) dst[i] = f2bf(src[i]);
}

// -------------------------------------------------- bf16 WMMA GEMM, C = A*B^T
// A: MxK row-major bf16 (lda). Bm: NxK row-major bf16 (ldb) => B[k][n]=Bm[n*ldb+k].
// C: MxN row-major f32 (ldc). Each wave computes a 16x64 strip (4 WMMA tiles).
union Frag16 { v16bf v; uint4 u[2]; };

__global__ __launch_bounds__(256) void gemm_bf16_nt(
    const unsigned short* __restrict__ A, const unsigned short* __restrict__ Bm,
    float* __restrict__ C, int M, int N, int K, int lda, int ldb, int ldc) {
  const int lane = threadIdx.x & 31;
  const int wave = threadIdx.x >> 5;
  const int gw   = blockIdx.x * 8 + wave;
  const int ngrp_cnt = N >> 6;                  // 64-wide N groups
  const int mt = gw / ngrp_cnt;
  const int ng = gw % ngrp_cnt;
  if (mt >= (M >> 4)) return;                   // wave-uniform: EXEC stays full
  const int m0 = mt << 4;
  const int n0 = ng << 6;
  const int lhalf = lane >> 4;                  // 0: lanes 0-15, 1: lanes 16-31
  const int lrow  = lane & 15;

  const unsigned short* arow  = A  + (size_t)(m0 + lrow) * lda;
  const unsigned short* brow0 = Bm + (size_t)(n0 + lrow) * ldb;

  v8f acc0 = (v8f)0.0f, acc1 = (v8f)0.0f, acc2 = (v8f)0.0f, acc3 = (v8f)0.0f;

  for (int k0 = 0; k0 < K; k0 += 32) {
    Frag16 fa;
    // A 16x32 bf16 layout: lanes0-15 K0-15, lanes16-31 K16-31 (matches B)
    fa.u[0] = *(const uint4*)(arow + k0 + lhalf * 16);
    fa.u[1] = *(const uint4*)(arow + k0 + lhalf * 16 + 8);
    __builtin_prefetch((const void*)(arow + k0 + 128), 0, 0);

    Frag16 fb;
    // B 32x16 bf16 layout: lanes0-15 K0-15, lanes16-31 K16-31 (K-contiguous rows)
    fb.u[0] = *(const uint4*)(brow0 + k0 + lhalf * 16);
    fb.u[1] = *(const uint4*)(brow0 + k0 + lhalf * 16 + 8);
    acc0 = __builtin_amdgcn_wmma_f32_16x16x32_bf16(false, fa.v, false, fb.v,
                                                   (short)0, acc0, false, false);
    fb.u[0] = *(const uint4*)(brow0 + (size_t)16 * ldb + k0 + lhalf * 16);
    fb.u[1] = *(const uint4*)(brow0 + (size_t)16 * ldb + k0 + lhalf * 16 + 8);
    acc1 = __builtin_amdgcn_wmma_f32_16x16x32_bf16(false, fa.v, false, fb.v,
                                                   (short)0, acc1, false, false);
    fb.u[0] = *(const uint4*)(brow0 + (size_t)32 * ldb + k0 + lhalf * 16);
    fb.u[1] = *(const uint4*)(brow0 + (size_t)32 * ldb + k0 + lhalf * 16 + 8);
    acc2 = __builtin_amdgcn_wmma_f32_16x16x32_bf16(false, fa.v, false, fb.v,
                                                   (short)0, acc2, false, false);
    fb.u[0] = *(const uint4*)(brow0 + (size_t)48 * ldb + k0 + lhalf * 16);
    fb.u[1] = *(const uint4*)(brow0 + (size_t)48 * ldb + k0 + lhalf * 16 + 8);
    acc3 = __builtin_amdgcn_wmma_f32_16x16x32_bf16(false, fa.v, false, fb.v,
                                                   (short)0, acc3, false, false);
  }

  // C/D layout: VGPR r -> row m0 + r + 8*lhalf, col n0 + 16*t + lrow
  float* crow = C + (size_t)(m0 + lhalf * 8) * ldc;
  const int c0 = n0 + lrow;
#pragma unroll
  for (int r = 0; r < 8; ++r) {
    crow[(size_t)r * ldc + c0]      = acc0[r];
    crow[(size_t)r * ldc + c0 + 16] = acc1[r];
    crow[(size_t)r * ldc + c0 + 32] = acc2[r];
    crow[(size_t)r * ldc + c0 + 48] = acc3[r];
  }
}

// ------------------------- causal conv(KW=4) + SiLU + q/k norm + g/beta (f32)
__global__ __launch_bounds__(256) void conv_silu_prep(
    const float* __restrict__ Y1, const float* __restrict__ conv_w,
    const float* __restrict__ A_log, const float* __restrict__ dt_bias,
    float* __restrict__ qbuf, float* __restrict__ kbuf, float* __restrict__ vbuf,
    float* __restrict__ gbuf, float* __restrict__ betabuf) {
  __shared__ float smem[CI_];
  __shared__ float hscale[32];
  const int b = blockIdx.x / S_;
  const int s = blockIdx.x % S_;
  const int tid = threadIdx.x;
  const size_t rowbase = (size_t)(b * S_ + s) * LDN_;

  for (int c = tid; c < CI_; c += 256) {
    float acc = 0.f;
#pragma unroll
    for (int j = 0; j < KW_; ++j) {
      int sp = s - (KW_ - 1) + j;
      if (sp >= 0) acc += conv_w[c * KW_ + j] * Y1[(size_t)(b * S_ + sp) * LDN_ + c];
    }
    smem[c] = acc * sigmoidf_(acc);             // SiLU
  }
  __syncthreads();
  if (tid < 32) {                               // 16 q-heads then 16 k-heads, each 128 wide
    float sum = 0.f;
    const int base = tid * DK_;
    for (int i = 0; i < DK_; ++i) { float v = smem[base + i]; sum += v * v; }
    float sc = rsqrtf(sum + 1e-6f);
    if (tid < 16) sc *= 0.08838834764831845f;   // * DK^-0.5 for q
    hscale[tid] = sc;
  }
  __syncthreads();
  const size_t qk_base = (size_t)(b * S_ + s) * QKD_;
  for (int i = tid; i < QKD_; i += 256) {
    qbuf[qk_base + i] = smem[i] * hscale[i >> 7];
    kbuf[qk_base + i] = smem[QKD_ + i] * hscale[16 + (i >> 7)];
  }
  const size_t v_base = (size_t)(b * S_ + s) * VD_;
  for (int i = tid; i < VD_; i += 256) vbuf[v_base + i] = smem[2 * QKD_ + i];
  if (tid < HV_) {
    float av = Y1[rowbase + CI_ + VD_ + tid];
    float bv = Y1[rowbase + CI_ + VD_ + HV_ + tid];
    float xg = av + dt_bias[tid];
    float sp = (xg > 20.f) ? xg : log1pf(__expf(xg));   // softplus
    gbuf[(size_t)(b * S_ + s) * HV_ + tid] = -__expf(A_log[tid]) * sp;
    betabuf[(size_t)(b * S_ + s) * HV_ + tid] = sigmoidf_(bv);
  }
}

// ---------------------------------------------------- conv_state = last KW cols
__global__ __launch_bounds__(256) void conv_state_copy(const float* __restrict__ Y1,
                                                       float* __restrict__ cs) {
  int idx = blockIdx.x * 256 + threadIdx.x;
  if (idx >= B_ * CI_ * KW_) return;
  int b = idx / (CI_ * KW_);
  int r = idx % (CI_ * KW_);
  int c = r / KW_, j = r % KW_;
  cs[idx] = Y1[(size_t)(b * S_ + (S_ - KW_) + j) * LDN_ + c];
}

// --------------------------- sequential gated delta-rule scan, state in regs.
// Double-buffered q/k/v staged to LDS with GLOBAL_LOAD_ASYNC_TO_LDS_B128 (ASYNCcnt),
// cross-thread reductions via wave32 shfl_xor (partner = tid^1): 1 barrier/step.
__global__ __launch_bounds__(256) void scan_kernel(
    const float* __restrict__ qbuf, const float* __restrict__ kbuf,
    const float* __restrict__ vbuf, const float* __restrict__ gbuf,
    const float* __restrict__ betabuf, float* __restrict__ ybuf,
    float* __restrict__ state_out) {
  const int bh = blockIdx.x;                    // b*HV + h
  const int b = bh / HV_, h = bh % HV_;
  const int hq = h >> 1;                        // GQA: g_rep = 2
  const int tid = threadIdx.x;
  const int vrow = tid >> 1;                    // DV row owned by this thread
  const int khalf = tid & 1;                    // which 64-wide K half
  const int kbase = khalf * 64;
  float st[64];
#pragma unroll
  for (int j = 0; j < 64; ++j) st[j] = 0.f;

  __shared__ __align__(16) float sq[2][DK_];
  __shared__ __align__(16) float sk[2][DK_];
  __shared__ __align__(16) float sv[2][DV_];

  // stage step s's q/k/v into LDS buffer nb: 3 waves issue one b128 async each
  auto stage = [&](int s, int nb) {
    const size_t qo = ((size_t)(b * S_ + s) * HK_ + hq) * DK_;
    const size_t vo = ((size_t)(b * S_ + s) * HV_ + h) * DV_;
    if (tid < 32)       async_ld_b128(ldsoff(&sq[nb][tid * 4]),        qbuf + qo + tid * 4);
    else if (tid < 64)  async_ld_b128(ldsoff(&sk[nb][(tid - 32) * 4]), kbuf + qo + (tid - 32) * 4);
    else if (tid < 96)  async_ld_b128(ldsoff(&sv[nb][(tid - 64) * 4]), vbuf + vo + (tid - 64) * 4);
  };

  stage(0, 0);
  wait_async0();
  __syncthreads();

  for (int s = 0; s < S_; ++s) {
    const int cur = s & 1;
    if (s + 1 < S_) stage(s + 1, cur ^ 1);      // prefetch next step behind compute

    // per-step scalars: uniform address -> broadcast load, no LDS staging
    const float eg  = __expf(gbuf[(size_t)(b * S_ + s) * HV_ + h]);
    const float bta = betabuf[(size_t)(b * S_ + s) * HV_ + h];

    float acc = 0.f;
#pragma unroll
    for (int j = 0; j < 64; ++j) { st[j] *= eg; acc += st[j] * sk[cur][kbase + j]; }
    const float kv = acc + __shfl_xor(acc, 1, 32);            // partner tid^1, same wave
    const float d  = (sv[cur][vrow] - kv) * bta;

    float oacc = 0.f;
#pragma unroll
    for (int j = 0; j < 64; ++j) { st[j] += sk[cur][kbase + j] * d; oacc += st[j] * sq[cur][kbase + j]; }
    const float o = oacc + __shfl_xor(oacc, 1, 32);
    if (khalf == 0)
      ybuf[((size_t)(b * S_ + s) * HV_ + h) * DV_ + vrow] = o;

    wait_async0();                              // next-step LDS data landed
    __syncthreads();                            // ... and visible to all waves
  }

  const size_t so = ((size_t)bh * DV_ + vrow) * DK_ + kbase;
#pragma unroll
  for (int j = 0; j < 64; ++j) state_out[so + j] = st[j];
}

// -------------------------------- RMS norm + SiLU(z) gate, emit bf16 for GEMM2
__global__ __launch_bounds__(128) void norm_gate(
    const float* __restrict__ ybuf, const float* __restrict__ Y1,
    const float* __restrict__ norm_w, unsigned short* __restrict__ ynb) {
  const int bs = blockIdx.x / HV_;
  const int h  = blockIdx.x % HV_;
  const int tid = threadIdx.x;
  __shared__ float red[128];
  const size_t yoff = ((size_t)bs * HV_ + h) * DV_;
  float y = ybuf[yoff + tid];
  red[tid] = y * y;
  __syncthreads();
  for (int off = 64; off > 0; off >>= 1) {
    if (tid < off) red[tid] += red[tid + off];
    __syncthreads();
  }
  float rms = rsqrtf(red[0] * (1.0f / DV_) + 1e-6f);
  float z = Y1[(size_t)bs * LDN_ + CI_ + h * DV_ + tid];
  float yn = norm_w[tid] * (y * rms) * (z * sigmoidf_(z));
  ynb[(size_t)bs * VD_ + h * DV_ + tid] = f2bf(yn);
}

// ---------------------------------------------------------------------- launch
extern "C" void kernel_launch(void* const* d_in, const int* in_sizes, int n_in,
                              void* d_out, int out_size, void* d_ws, size_t ws_size,
                              hipStream_t stream) {
  (void)in_sizes; (void)n_in; (void)out_size; (void)ws_size;
  const float* x      = (const float*)d_in[0];
  const float* W_qkv  = (const float*)d_in[1];
  const float* W_z    = (const float*)d_in[2];
  const float* W_a    = (const float*)d_in[3];
  const float* W_b    = (const float*)d_in[4];
  const float* conv_w = (const float*)d_in[5];
  const float* A_log  = (const float*)d_in[6];
  const float* dt_b   = (const float*)d_in[7];
  const float* norm_w = (const float*)d_in[8];
  const float* W_out  = (const float*)d_in[9];
  float* out = (float*)d_out;

  char* ws = (char*)d_ws;
  size_t off = 0;
  auto take = [&](size_t bytes) {
    char* p = ws + off;
    off += (bytes + 255) & ~(size_t)255;
    return p;
  };
  unsigned short* xb    = (unsigned short*)take((size_t)MROWS_ * H_ * 2);
  unsigned short* wcat  = (unsigned short*)take((size_t)LDN_ * H_ * 2);
  unsigned short* woutb = (unsigned short*)take((size_t)H_ * VD_ * 2);
  float* Y1   = (float*)take((size_t)MROWS_ * LDN_ * 4);
  float* qbuf = (float*)take((size_t)MROWS_ * QKD_ * 4);
  float* kbuf = (float*)take((size_t)MROWS_ * QKD_ * 4);
  float* vbuf = (float*)take((size_t)MROWS_ * VD_ * 4);
  float* gbuf = (float*)take((size_t)MROWS_ * HV_ * 4);
  float* bbuf = (float*)take((size_t)MROWS_ * HV_ * 4);
  float* ybuf = (float*)take((size_t)MROWS_ * VD_ * 4);
  unsigned short* ynb = (unsigned short*)take((size_t)MROWS_ * VD_ * 2);

  // 1) bf16 conversions (weights end up L2-resident: ~50 MB of 192 MB)
  cvt_f32_bf16<<<1024, 256, 0, stream>>>(x, xb, (long long)MROWS_ * H_);
  cvt_f32_bf16<<<1024, 256, 0, stream>>>(W_qkv, wcat, (long long)CI_ * H_);
  cvt_f32_bf16<<<1024, 256, 0, stream>>>(W_z, wcat + (size_t)CI_ * H_, (long long)VD_ * H_);
  cvt_f32_bf16<<<64, 256, 0, stream>>>(W_a, wcat + (size_t)(CI_ + VD_) * H_, (long long)HV_ * H_);
  cvt_f32_bf16<<<64, 256, 0, stream>>>(W_b, wcat + (size_t)(CI_ + VD_ + HV_) * H_, (long long)HV_ * H_);
  cvt_f32_bf16<<<1024, 256, 0, stream>>>(W_out, woutb, (long long)H_ * VD_);

  // 2) fused projection GEMM: Y1 = x @ [W_qkv;W_z;W_a;W_b]^T  (4096 x 12352, K=2048)
  {
    int waves = (MROWS_ / 16) * (LDN_ / 64);     // 49408, divisible by 8
    gemm_bf16_nt<<<waves / 8, 256, 0, stream>>>(xb, wcat, Y1, MROWS_, LDN_, H_, H_, H_, LDN_);
  }

  // 3) conv + activations + head norms
  conv_silu_prep<<<B_ * S_, 256, 0, stream>>>(Y1, conv_w, A_log, dt_b,
                                              qbuf, kbuf, vbuf, gbuf, bbuf);
  conv_state_copy<<<(B_ * CI_ * KW_ + 255) / 256, 256, 0, stream>>>(
      Y1, out + (size_t)MROWS_ * H_);

  // 4) sequential delta-rule scan; final state to output region 3
  scan_kernel<<<B_ * HV_, 256, 0, stream>>>(
      qbuf, kbuf, vbuf, gbuf, bbuf, ybuf,
      out + (size_t)MROWS_ * H_ + (size_t)B_ * CI_ * KW_);

  // 5) RMS norm + gate, then output GEMM: out = yn @ W_out^T (4096 x 2048, K=4096)
  norm_gate<<<MROWS_ * HV_, 128, 0, stream>>>(ybuf, Y1, norm_w, ynb);
  {
    int waves = (MROWS_ / 16) * (H_ / 64);       // 8192, divisible by 8
    gemm_bf16_nt<<<waves / 8, 256, 0, stream>>>(ynb, woutb, out, MROWS_, H_, VD_, VD_, VD_, H_);
  }
}
